// GRU_68367289417741
// MI455X (gfx1250) — compile-verified, matrix operand order checked by reading, not run on hardware
//
#include <hip/hip_runtime.h>
#include <hip/hip_bf16.h>
#include <math.h>

// Problem constants (from reference)
#define V_ 10000
#define E_ 512
#define H_ 1024
#define L_ 2
#define T_ 128
#define B_ 64
#define TB_ (T_ * B_)

// LDS K-panel for the big GEMMs
#define KB_ 32
#define KBP_ 40   // padded row stride (bf16 elems): 80B -> 20 banks, conflict-free

// CDNA5 async global->LDS path (guarded: falls back to load+ds_store if the
// toolchain doesn't declare the gfx1250 async builtins).
#if defined(__has_builtin)
#if __has_builtin(__builtin_amdgcn_global_load_async_to_lds_b128)
#define HAVE_ASYNC_LDS 1
#endif
#endif

typedef __bf16 bf16;
typedef bf16  v8bf  __attribute__((ext_vector_type(8)));
typedef bf16  v16bf __attribute__((ext_vector_type(16)));
typedef float v8f   __attribute__((ext_vector_type(8)));
typedef int   v4i   __attribute__((ext_vector_type(4)));

typedef __attribute__((address_space(1))) v4i* gv4i_p;   // global int4*
typedef __attribute__((address_space(3))) v4i* lv4i_p;   // LDS int4*

// ---------------------------------------------------------------- helpers ---

__device__ __forceinline__ bf16 f2bf(float f) {
  unsigned u = __builtin_bit_cast(unsigned, f);
  u += 0x7FFFu + ((u >> 16) & 1u);          // round-to-nearest-even
  unsigned short h = (unsigned short)(u >> 16);
  return __builtin_bit_cast(bf16, h);
}

__device__ __forceinline__ void wait_async_lds() {
#if defined(HAVE_ASYNC_LDS)
#if __has_builtin(__builtin_amdgcn_s_wait_asynccnt)
  __builtin_amdgcn_s_wait_asynccnt(0);
#else
  asm volatile("s_wait_asynccnt 0x0" ::: "memory");
#endif
#endif
}

// Load one 16x32 bf16 WMMA fragment for this lane from a row-major source
// (global or LDS). p = row base pointer (row already selected per-lane).
// ISA layout (wave32): lanes 0-15 hold K = hlf*8 + [0..7] (VGPR0..3) and
// K = 16 + hlf*8 + [0..7] (VGPR4..7), where hlf = lane>>4.
__device__ __forceinline__ v16bf ld_frag(const bf16* __restrict__ p, int k0, int hlf) {
  const v8bf* q = (const v8bf*)(p + k0 + hlf * 8);
  v8bf lo = q[0];        // K = k0 + hlf*8 + 0..7
  v8bf hi = q[2];        // K = k0 + 16 + hlf*8 + 0..7
  return __builtin_shufflevector(lo, hi, 0, 1, 2, 3, 4, 5, 6, 7,
                                 8, 9, 10, 11, 12, 13, 14, 15);
}

__device__ __forceinline__ v8f wmma_bf16(v16bf a, v16bf b, v8f c) {
  return __builtin_amdgcn_wmma_f32_16x16x32_bf16(
      /*neg_a=*/false, a, /*neg_b=*/false, b,
      /*c_mod=*/(short)0, c, /*reuse_a=*/false, /*reuse_b=*/false);
}

// Direct-from-global accumulate C[16x64] += A[16xK] * W[64xK]^T (cell kernels).
__device__ __forceinline__ void acc_loop(const bf16* __restrict__ Arow,
                                         const bf16* __restrict__ W0,
                                         const bf16* __restrict__ W1,
                                         const bf16* __restrict__ W2,
                                         const bf16* __restrict__ W3,
                                         int K, int hlf, v8f c[4]) {
  for (int k0 = 0; k0 < K; k0 += 32) {
    v16bf a = ld_frag(Arow, k0, hlf);
    c[0] = wmma_bf16(a, ld_frag(W0, k0, hlf), c[0]);
    c[1] = wmma_bf16(a, ld_frag(W1, k0, hlf), c[1]);
    c[2] = wmma_bf16(a, ld_frag(W2, k0, hlf), c[2]);
    c[3] = wmma_bf16(a, ld_frag(W3, k0, hlf), c[3]);
  }
}

__device__ __forceinline__ float sigmoidf_(float x) {
  return 1.0f / (1.0f + __expf(-x));
}

// ---------------------------------------------------------------- kernels ---

__global__ void k_f32_to_bf16(const float* __restrict__ src,
                              bf16* __restrict__ dst, int n) {
  int i = blockIdx.x * blockDim.x + threadIdx.x;
  int stride = gridDim.x * blockDim.x;
  for (; i < n; i += stride) dst[i] = f2bf(src[i]);
}

__global__ void k_embed(const int* __restrict__ inp,
                        const float* __restrict__ emb,
                        bf16* __restrict__ X) {
  int idx = blockIdx.x * blockDim.x + threadIdx.x;   // TB_*E_ threads
  if (idx >= TB_ * E_) return;
  int tb = idx >> 9;            // / E_ (512)
  int e  = idx & (E_ - 1);
  X[idx] = f2bf(emb[(size_t)inp[tb] * E_ + e]);
}

// Big GEMM: C[M,N](f32) = A_bf[M,K] @ W_bf[N,K]^T + bias[N]
// Workgroup: 256 threads = 8 waves. WG tile 128x128 (wave tile 32x64,
// 8 WMMA accumulators). K staged through LDS in double-buffered KB_=32
// panels; fragments read from LDS (padded stride -> bank-conflict-free).
// Staging uses GLOBAL_LOAD_ASYNC_TO_LDS_B128 (ASYNCcnt) when available.
// Requires: M multiple of 128 (gridDim.y*128), K multiple of KB_.
// N may be ragged (loads clamped, stores guarded).
__global__ __launch_bounds__(256) void
k_gemm_nt_bias(const bf16* __restrict__ A, int lda,
               const bf16* __restrict__ W, int ldw,
               const float* __restrict__ bias,
               float* __restrict__ C, long ldc, int N, int K) {
  __shared__ __attribute__((aligned(16))) bf16 sA[2][128 * KBP_];
  __shared__ __attribute__((aligned(16))) bf16 sB[2][128 * KBP_];

  const int tid  = threadIdx.x;
  const int wave = tid >> 5;
  const int lane = tid & 31;
  const int hlf  = lane >> 4;
  const int l16  = lane & 15;
  const int mBase = blockIdx.y * 128;
  const int nBase = blockIdx.x * 128;
  const int m0 = (wave & 3) * 32;     // wave row offset within WG tile
  const int n0 = (wave >> 2) * 64;    // wave col offset within WG tile

  // Cooperative stage of one 128x32 K-panel of A and W into LDS.
  // 512 b128 transfers per operand / 256 threads = 2 each.
  auto stage = [&](int buf, int kb) {
#pragma unroll
    for (int i = 0; i < 2; ++i) {
      int idx = tid + i * 256;        // 0..511
      int row = idx >> 2;             // 0..127
      int ch  = (idx & 3) * 8;        // 0,8,16,24
      const bf16* ga = A + (size_t)(mBase + row) * lda + kb + ch;
      bf16*       la = sA[buf] + row * KBP_ + ch;
      int wr = nBase + row;
      if (wr > N - 1) wr = N - 1;     // clamp OOB W rows
      const bf16* gb = W + (size_t)wr * ldw + kb + ch;
      bf16*       lb = sB[buf] + row * KBP_ + ch;
#if defined(HAVE_ASYNC_LDS)
      __builtin_amdgcn_global_load_async_to_lds_b128(
          (gv4i_p)ga, (lv4i_p)la, 0, 0);
      __builtin_amdgcn_global_load_async_to_lds_b128(
          (gv4i_p)gb, (lv4i_p)lb, 0, 0);
#else
      *(v8bf*)la = *(const v8bf*)ga;
      *(v8bf*)lb = *(const v8bf*)gb;
#endif
    }
  };

  v8f c[8] = {};
  stage(0, 0);
  int buf = 0;
  for (int kb = 0; kb < K; kb += KB_) {
    wait_async_lds();                         // my async transfers landed in LDS
    __syncthreads();                          // staged panel visible to all
    if (kb + KB_ < K) stage(buf ^ 1, kb + KB_);   // prefetch next panel
    v16bf a0 = ld_frag(sA[buf] + (m0 + l16) * KBP_, 0, hlf);
    v16bf a1 = ld_frag(sA[buf] + (m0 + 16 + l16) * KBP_, 0, hlf);
#pragma unroll
    for (int j = 0; j < 4; ++j) {
      v16bf b = ld_frag(sB[buf] + (n0 + j * 16 + l16) * KBP_, 0, hlf);
      c[j]     = wmma_bf16(a0, b, c[j]);
      c[4 + j] = wmma_bf16(a1, b, c[4 + j]);
    }
    buf ^= 1;
  }

#pragma unroll
  for (int j = 0; j < 4; ++j) {
    int col = nBase + n0 + j * 16 + l16;
    if (col < N) {
      float bb = bias ? bias[col] : 0.0f;
#pragma unroll
      for (int r = 0; r < 8; ++r) {
        int row0 = mBase + m0 + hlf * 8 + r;
        int row1 = row0 + 16;
        C[(size_t)row0 * ldc + col] = c[j][r] + bb;
        C[(size_t)row1 * ldc + col] = c[4 + j][r] + bb;
      }
    }
  }
}

// GRU r/z gates for one timestep, one layer. Output [B_, H_].
// gate = blockIdx.y (0=r, 1=z). Grid: (H_/128, 2).
// acc = h@U^T (+ x@W^T for layer1); v = acc (+gx)(+bias); s = sigmoid(v)
// gate r: rhout = bf16(s * hprev);  gate z: zout = s
__global__ __launch_bounds__(256) void
k_cell_rz(const bf16* __restrict__ hbf,
          const bf16* __restrict__ Ur, const bf16* __restrict__ Uz,
          const bf16* __restrict__ xbf,                         // null for layer0
          const bf16* __restrict__ Wr, const bf16* __restrict__ Wz,
          const float* __restrict__ gxr, const float* __restrict__ gxz,
          const float* __restrict__ br, const float* __restrict__ bz,
          const float* __restrict__ hprev,
          bf16* __restrict__ rhout, float* __restrict__ zout) {
  const int gate = blockIdx.y;
  const bf16*  U    = gate ? Uz  : Ur;
  const bf16*  W    = gate ? Wz  : Wr;
  const float* gx   = gate ? gxz : gxr;
  const float* bias = gate ? bz  : br;

  const int tid  = threadIdx.x;
  const int wave = tid >> 5;
  const int lane = tid & 31;
  const int hlf  = lane >> 4;
  const int l16  = lane & 15;
  const int m0   = (wave & 3) * 16;                  // B_ = 64 rows
  const int n0   = blockIdx.x * 128 + (wave >> 2) * 64;

  v8f c[4] = {};
  {
    const bf16* Arow = hbf + (size_t)(m0 + l16) * H_;
    const bf16* u0 = U + (size_t)(n0 + 0 * 16 + l16) * H_;
    const bf16* u1 = U + (size_t)(n0 + 1 * 16 + l16) * H_;
    const bf16* u2 = U + (size_t)(n0 + 2 * 16 + l16) * H_;
    const bf16* u3 = U + (size_t)(n0 + 3 * 16 + l16) * H_;
    acc_loop(Arow, u0, u1, u2, u3, H_, hlf, c);
  }
  if (xbf) {                                          // layer1: + x @ W^T
    const bf16* Arow = xbf + (size_t)(m0 + l16) * H_;
    const bf16* w0 = W + (size_t)(n0 + 0 * 16 + l16) * H_;
    const bf16* w1 = W + (size_t)(n0 + 1 * 16 + l16) * H_;
    const bf16* w2 = W + (size_t)(n0 + 2 * 16 + l16) * H_;
    const bf16* w3 = W + (size_t)(n0 + 3 * 16 + l16) * H_;
    acc_loop(Arow, w0, w1, w2, w3, H_, hlf, c);
  }

#pragma unroll
  for (int j = 0; j < 4; ++j) {
    int col = n0 + j * 16 + l16;
#pragma unroll
    for (int r = 0; r < 8; ++r) {
      int row = m0 + hlf * 8 + r;
      size_t idx = (size_t)row * H_ + col;
      float v = c[j][r];
      if (gx)   v += gx[idx];
      if (bias) v += bias[col];
      float s = sigmoidf_(v);
      if (gate == 0) rhout[idx] = f2bf(s * hprev[idx]);
      else           zout[idx]  = s;
    }
  }
}

// GRU h-tilde + state update. acc = rh@Uh^T (+ x@Wh^T);
// ht = tanh(acc (+gx)(+bias)); hn = (1-z)*hprev + z*ht
__global__ __launch_bounds__(256) void
k_cell_h(const bf16* __restrict__ rhbf, const bf16* __restrict__ Uh,
         const bf16* __restrict__ xbf,  const bf16* __restrict__ Wh,
         const float* __restrict__ gxh, const float* __restrict__ bh,
         const float* __restrict__ zbuf, const float* __restrict__ hprev,
         float* __restrict__ houtf, bf16* __restrict__ houtbf,
         bf16* __restrict__ top) {
  const int tid  = threadIdx.x;
  const int wave = tid >> 5;
  const int lane = tid & 31;
  const int hlf  = lane >> 4;
  const int l16  = lane & 15;
  const int m0   = (wave & 3) * 16;
  const int n0   = blockIdx.x * 128 + (wave >> 2) * 64;

  v8f c[4] = {};
  {
    const bf16* Arow = rhbf + (size_t)(m0 + l16) * H_;
    const bf16* u0 = Uh + (size_t)(n0 + 0 * 16 + l16) * H_;
    const bf16* u1 = Uh + (size_t)(n0 + 1 * 16 + l16) * H_;
    const bf16* u2 = Uh + (size_t)(n0 + 2 * 16 + l16) * H_;
    const bf16* u3 = Uh + (size_t)(n0 + 3 * 16 + l16) * H_;
    acc_loop(Arow, u0, u1, u2, u3, H_, hlf, c);
  }
  if (xbf) {
    const bf16* Arow = xbf + (size_t)(m0 + l16) * H_;
    const bf16* w0 = Wh + (size_t)(n0 + 0 * 16 + l16) * H_;
    const bf16* w1 = Wh + (size_t)(n0 + 1 * 16 + l16) * H_;
    const bf16* w2 = Wh + (size_t)(n0 + 2 * 16 + l16) * H_;
    const bf16* w3 = Wh + (size_t)(n0 + 3 * 16 + l16) * H_;
    acc_loop(Arow, w0, w1, w2, w3, H_, hlf, c);
  }

#pragma unroll
  for (int j = 0; j < 4; ++j) {
    int col = n0 + j * 16 + l16;
#pragma unroll
    for (int r = 0; r < 8; ++r) {
      int row = m0 + hlf * 8 + r;
      size_t idx = (size_t)row * H_ + col;
      float v = c[j][r];
      if (gxh) v += gxh[idx];
      if (bh)  v += bh[col];
      float ht = tanhf(v);
      float z  = zbuf[idx];
      float hn = (1.0f - z) * hprev[idx] + z * ht;
      houtf[idx]  = hn;
      houtbf[idx] = f2bf(hn);
      if (top) top[idx] = f2bf(hn);
    }
  }
}

// ----------------------------------------------------------------- launch ---

extern "C" void kernel_launch(void* const* d_in, const int* in_sizes, int n_in,
                              void* d_out, int out_size, void* d_ws, size_t ws_size,
                              hipStream_t stream) {
  const int*   inp    = (const int*)d_in[0];
  const float* hidden = (const float*)d_in[1];
  const float* emb    = (const float*)d_in[2];
  const float* Wr0 = (const float*)d_in[3];
  const float* br0 = (const float*)d_in[4];
  const float* Wz0 = (const float*)d_in[5];
  const float* bz0 = (const float*)d_in[6];
  const float* Wh0 = (const float*)d_in[7];
  const float* bh0 = (const float*)d_in[8];
  const float* Ur0 = (const float*)d_in[9];
  const float* Uz0 = (const float*)d_in[10];
  const float* Uh0 = (const float*)d_in[11];
  const float* Wr1 = (const float*)d_in[12];
  const float* br1 = (const float*)d_in[13];
  const float* Wz1 = (const float*)d_in[14];
  const float* bz1 = (const float*)d_in[15];
  const float* Wh1 = (const float*)d_in[16];
  const float* bh1 = (const float*)d_in[17];
  const float* Ur1 = (const float*)d_in[18];
  const float* Uz1 = (const float*)d_in[19];
  const float* Uh1 = (const float*)d_in[20];
  const float* Wout = (const float*)d_in[21];
  const float* bout = (const float*)d_in[22];

  float* logits = (float*)d_out;                       // [T,B,V]
  float* hstate = logits + (size_t)TB_ * V_;           // [L,B,H] (= h_final)

  // ---- workspace carve (bf16 region first, then f32 region) ----
  bf16* p = (bf16*)d_ws;
  bf16* bWr0 = p; p += (size_t)H_ * E_;
  bf16* bWz0 = p; p += (size_t)H_ * E_;
  bf16* bWh0 = p; p += (size_t)H_ * E_;
  bf16* bUr0 = p; p += (size_t)H_ * H_;
  bf16* bUz0 = p; p += (size_t)H_ * H_;
  bf16* bUh0 = p; p += (size_t)H_ * H_;
  bf16* bWr1 = p; p += (size_t)H_ * H_;
  bf16* bWz1 = p; p += (size_t)H_ * H_;
  bf16* bWh1 = p; p += (size_t)H_ * H_;
  bf16* bUr1 = p; p += (size_t)H_ * H_;
  bf16* bUz1 = p; p += (size_t)H_ * H_;
  bf16* bUh1 = p; p += (size_t)H_ * H_;
  bf16* bWout = p; p += (size_t)V_ * H_;
  bf16* Xbf   = p; p += (size_t)TB_ * E_;
  bf16* hbf   = p; p += (size_t)L_ * B_ * H_;          // hbf0, hbf1
  bf16* rhbf  = p; p += (size_t)B_ * H_;
  bf16* topbf = p; p += (size_t)TB_ * H_;
  float* fp = (float*)p;                                // 16B aligned by construction
  float* gxr  = fp; fp += (size_t)TB_ * H_;
  float* gxz  = fp; fp += (size_t)TB_ * H_;
  float* gxh  = fp; fp += (size_t)TB_ * H_;
  float* zbuf = fp; fp += (size_t)B_ * H_;

  // ---- weight / state conversion to bf16 ----
  auto cvt = [&](const float* s, bf16* d, size_t n) {
    int blocks = (int)((n + 255) / 256);
    k_f32_to_bf16<<<blocks, 256, 0, stream>>>(s, d, (int)n);
  };
  cvt(Wr0, bWr0, (size_t)H_ * E_);
  cvt(Wz0, bWz0, (size_t)H_ * E_);
  cvt(Wh0, bWh0, (size_t)H_ * E_);
  cvt(Ur0, bUr0, (size_t)H_ * H_);
  cvt(Uz0, bUz0, (size_t)H_ * H_);
  cvt(Uh0, bUh0, (size_t)H_ * H_);
  cvt(Wr1, bWr1, (size_t)H_ * H_);
  cvt(Wz1, bWz1, (size_t)H_ * H_);
  cvt(Wh1, bWh1, (size_t)H_ * H_);
  cvt(Ur1, bUr1, (size_t)H_ * H_);
  cvt(Uz1, bUz1, (size_t)H_ * H_);
  cvt(Uh1, bUh1, (size_t)H_ * H_);
  cvt(Wout, bWout, (size_t)V_ * H_);
  cvt(hidden, hbf, (size_t)L_ * B_ * H_);
  (void)hipMemcpyAsync(hstate, hidden, sizeof(float) * L_ * B_ * H_,
                       hipMemcpyDeviceToDevice, stream);

  // ---- embedding gather -> bf16 ----
  {
    int n = TB_ * E_;
    k_embed<<<(n + 255) / 256, 256, 0, stream>>>(inp, emb, Xbf);
  }

  // ---- batched layer-0 input projections (all T at once): gx = X@W^T + b ----
  {
    dim3 g(H_ / 128, TB_ / 128);   // (8, 64)
    k_gemm_nt_bias<<<g, 256, 0, stream>>>(Xbf, E_, bWr0, E_, br0, gxr, H_, H_, E_);
    k_gemm_nt_bias<<<g, 256, 0, stream>>>(Xbf, E_, bWz0, E_, bz0, gxz, H_, H_, E_);
    k_gemm_nt_bias<<<g, 256, 0, stream>>>(Xbf, E_, bWh0, E_, bh0, gxh, H_, H_, E_);
  }

  // ---- sequential recurrence ----
  bf16* hbf0 = hbf;
  bf16* hbf1 = hbf + (size_t)B_ * H_;
  float* h0f = hstate;
  float* h1f = hstate + (size_t)B_ * H_;
  dim3 gc(H_ / 128, 2);   // r,z gates
  dim3 gh(H_ / 128, 1);
  for (int t = 0; t < T_; ++t) {
    const float* gr = gxr + (size_t)t * B_ * H_;
    const float* gz = gxz + (size_t)t * B_ * H_;
    const float* gH = gxh + (size_t)t * B_ * H_;
    // layer 0 (input-side matmuls precomputed in gx*)
    k_cell_rz<<<gc, 256, 0, stream>>>(hbf0, bUr0, bUz0,
                                      nullptr, nullptr, nullptr,
                                      gr, gz, nullptr, nullptr,
                                      h0f, rhbf, zbuf);
    k_cell_h<<<gh, 256, 0, stream>>>(rhbf, bUh0, nullptr, nullptr,
                                     gH, nullptr, zbuf, h0f,
                                     h0f, hbf0, nullptr);
    // layer 1 (x = layer-0 new hidden; fused W+U accumulation)
    k_cell_rz<<<gc, 256, 0, stream>>>(hbf1, bUr1, bUz1,
                                      hbf0, bWr1, bWz1,
                                      nullptr, nullptr, br1, bz1,
                                      h1f, rhbf, zbuf);
    k_cell_h<<<gh, 256, 0, stream>>>(rhbf, bUh1, hbf0, bWh1,
                                     nullptr, bh1, zbuf, h1f,
                                     h1f, hbf1, topbf + (size_t)t * B_ * H_);
  }

  // ---- output projection: logits = top @ Wout^T + bout ----
  {
    dim3 g((V_ + 127) / 128, TB_ / 128);   // (79, 64)
    k_gemm_nt_bias<<<g, 256, 0, stream>>>(topbf, H_, bWout, H_, bout,
                                          logits, V_, V_, H_);
  }
}
